// DotAttention_1932735283232
// MI455X (gfx1250) — matrix operand order repaired and checked by reading, not compiled, not run on hardware
//
#include <hip/hip_runtime.h>
#include <hip/hip_bf16.h>

typedef __attribute__((ext_vector_type(16))) __bf16 v16bf;
typedef __attribute__((ext_vector_type(8)))  __bf16 v8bf;
typedef __attribute__((ext_vector_type(4)))  __bf16 v4bf;
typedef __attribute__((ext_vector_type(8)))  float  v8f;
typedef __attribute__((ext_vector_type(4)))  float  v4f;

#define NB   32      // batch
#define TE   2048    // encoder length
#define TD   512     // decoder length
#define DD   512     // model dim (K of projection)
#define NN   512     // n_state

// ---------------------------------------------------------------------------
// WMMA fragment helpers (wave32, v_wmma_f32_16x16x32_bf16)
// ---------------------------------------------------------------------------

// A-matrix 16x32 bf16 fragment from a 64x32 bf16 LDS tile (row-major, ld=32).
// lane L: row M = rbase + L%16, half = L/16; element i -> K = 16*half + i
static __device__ __forceinline__ v16bf lds_a_frag(const __bf16* lsA, int rbase, int lane) {
  const int m    = rbase + (lane & 15);
  const int half = lane >> 4;
  const __bf16* p = lsA + m * 32 + 16 * half;
  v8bf lo = *(const v8bf*)p;          // K = 16*half + 0..7
  v8bf hi = *(const v8bf*)(p + 8);    // K = 16*half + 8..15
  v16bf a;
#pragma unroll
  for (int i = 0; i < 8; ++i) { a[i] = lo[i]; a[8 + i] = hi[i]; }
  return a;
}

// B-matrix 32x16 bf16 fragment from a K-contiguous bf16 operand.
// lane L: col N = col0 + L%16, elements K = k0 + 16*(L/16) + i  (16 contiguous halves)
static __device__ __forceinline__ v16bf load_b_frag_bf16(const __bf16* __restrict__ base,
                                                         int ld, int col0, int k0, int lane) {
  const int n    = col0 + (lane & 15);
  const int half = lane >> 4;
  return *(const v16bf*)(base + (size_t)n * ld + k0 + 16 * half);
}

static __device__ __forceinline__ v8f wmma_bf16(v16bf a, v16bf b, v8f c) {
  // (neg_a, A, neg_b, B, c_mod, C, reuse_a, reuse_b)
  return __builtin_amdgcn_wmma_f32_16x16x32_bf16(false, a, false, b, (short)0, c,
                                                 false, false);
}

// C/D 16x16 f32 layout: lane L -> col N = L%16, rows M = 8*(L/16)+r in VGPR r.
static __device__ __forceinline__ void store_f32_tile(v8f c, float* __restrict__ dst,
                                                      int ldd, int row0, int col0, int lane) {
  const int n  = col0 + (lane & 15);
  const int mb = row0 + 8 * (lane >> 4);
#pragma unroll
  for (int r = 0; r < 8; ++r)
    dst[(size_t)(mb + r) * ldd + n] = c[r];
}

// Cooperative LDS A-tile fill: 64 rows x 32 K bf16, 256 threads x 16B.
// tid -> row = tid/4, kchunk = (tid%4)*8.
static __device__ __forceinline__ void fill_a_tile_bf16(__bf16* lsA,
                                                        const __bf16* __restrict__ src,
                                                        int ld, int row0, int k0, int tid) {
  const int row = tid >> 2;
  const int kk  = (tid & 3) << 3;
  v8bf y = *(const v8bf*)(src + (size_t)(row0 + row) * ld + k0 + kk);
  *(v8bf*)&lsA[row * 32 + kk] = y;
}

// Same, converting from an f32 source on the fly (32B load -> 16B store).
static __device__ __forceinline__ void fill_a_tile_f32(__bf16* lsA,
                                                       const float* __restrict__ src,
                                                       int ld, int row0, int k0, int tid) {
  const int row = tid >> 2;
  const int kk  = (tid & 3) << 3;
  const float* p = src + (size_t)(row0 + row) * ld + k0 + kk;
  v4f x0 = *(const v4f*)p;
  v4f x1 = *(const v4f*)(p + 4);
  v8bf y;
#pragma unroll
  for (int i = 0; i < 4; ++i) { y[i] = (__bf16)x0[i]; y[4 + i] = (__bf16)x1[i]; }
  *(v8bf*)&lsA[row * 32 + kk] = y;
}

// ---------------------------------------------------------------------------
// Kernel 0a: pack W^T into bf16  (Wt[n, d] = bf16(W[d, n]))
// ---------------------------------------------------------------------------
__global__ void __launch_bounds__(256) wt_pack_kernel(const float* __restrict__ W,
                                                      __bf16* __restrict__ Wt) {
  const int idx = blockIdx.x * 256 + threadIdx.x;   // 512*512 threads
  const int n = idx >> 9;
  const int d = idx & 511;
  Wt[(size_t)n * DD + d] = (__bf16)W[(size_t)d * NN + n];
}

// ---------------------------------------------------------------------------
// Kernel 0b: pack decoder states to bf16 (same layout), 4 elements/thread.
// ---------------------------------------------------------------------------
__global__ void __launch_bounds__(256) dec_pack_kernel(const float* __restrict__ dec,
                                                       __bf16* __restrict__ dec_bf) {
  const size_t base = ((size_t)blockIdx.x * 256 + threadIdx.x) * 4;
  v4f x = *(const v4f*)(dec + base);
  v4bf y;
#pragma unroll
  for (int i = 0; i < 4; ++i) y[i] = (__bf16)x[i];
  *(v4bf*)(dec_bf + base) = y;
}

// ---------------------------------------------------------------------------
// Kernel 1: enc_proj = enc @ W + b.
// Block = 8 waves = 64 e-rows x 512 n (full N); wave tile 64x64 (16 wmma/step).
// A (enc, f32->bf16) double-buffered in LDS; one barrier per K-step.
// Writes bf16 enc_proj [B,E,N] and enc_proj^T [B,N,E].
// ---------------------------------------------------------------------------
static __device__ __forceinline__ void store_proj_tile(v8f c, __bf16* __restrict__ ep,
                                                       __bf16* __restrict__ epT,
                                                       int row0, int col0, int lane) {
  const int n  = col0 + (lane & 15);
  const int mb = row0 + 8 * (lane >> 4);
#pragma unroll
  for (int r = 0; r < 8; ++r) {
    __bf16 v = (__bf16)c[r];
    ep[(size_t)(mb + r) * NN + n] = v;          // enc_proj[e, n]
    epT[(size_t)n * TE + (mb + r)] = v;         // enc_proj^T[n, e]
  }
}

__global__ void __launch_bounds__(256) proj_kernel(const float* __restrict__ enc,
                                                   const float* __restrict__ bias,
                                                   const __bf16* __restrict__ Wt,
                                                   __bf16* __restrict__ encp,
                                                   __bf16* __restrict__ encpT) {
  __shared__ __bf16 ls[2][64 * 32];
  const int tid  = threadIdx.x;
  const int lane = tid & 31;
  const int wv   = tid >> 5;                 // 0..7
  const int b    = blockIdx.x >> 5;          // / 32 m-tiles
  const int row0 = (blockIdx.x & 31) * 64;   // e
  const int col0 = wv * 64;                  // n band per wave
  const float* encB = enc + (size_t)b * TE * DD;

  v8f c[4][4];
#pragma unroll
  for (int j = 0; j < 4; ++j) {
    const float bv = bias[col0 + j * 16 + (lane & 15)];
#pragma unroll
    for (int mi = 0; mi < 4; ++mi)
#pragma unroll
      for (int i = 0; i < 8; ++i) c[mi][j][i] = bv;
  }

  fill_a_tile_f32(ls[0], encB, DD, row0, 0, tid);
  __syncthreads();
  for (int k0 = 0; k0 < DD; k0 += 32) {
    const int cur = (k0 >> 5) & 1;
    if (k0 + 32 < DD) fill_a_tile_f32(ls[cur ^ 1], encB, DD, row0, k0 + 32, tid);
    v16bf a0 = lds_a_frag(ls[cur], 0,  lane);
    v16bf a1 = lds_a_frag(ls[cur], 16, lane);
    v16bf a2 = lds_a_frag(ls[cur], 32, lane);
    v16bf a3 = lds_a_frag(ls[cur], 48, lane);
#pragma unroll
    for (int j = 0; j < 4; ++j) {
      v16bf bm = load_b_frag_bf16(Wt, DD, col0 + 16 * j, k0, lane);
      c[0][j] = wmma_bf16(a0, bm, c[0][j]);
      c[1][j] = wmma_bf16(a1, bm, c[1][j]);
      c[2][j] = wmma_bf16(a2, bm, c[2][j]);
      c[3][j] = wmma_bf16(a3, bm, c[3][j]);
    }
    __syncthreads();
  }

  __bf16* ep  = encp  + (size_t)b * TE * NN;
  __bf16* epT = encpT + (size_t)b * NN * TE;
#pragma unroll
  for (int mi = 0; mi < 4; ++mi)
#pragma unroll
    for (int j = 0; j < 4; ++j)
      store_proj_tile(c[mi][j], ep, epT, row0 + 16 * mi, col0 + 16 * j, lane);
}

// ---------------------------------------------------------------------------
// Kernel 2: score[b,t,e] = sum_n dec[b,t,n] * enc_proj[b,e,n].
// Block = 64 t-rows x 512 e-band (4 bands). A = dec_bf double-buffered in LDS.
// Raw scores written into the attention output slice.
// ---------------------------------------------------------------------------
__global__ void __launch_bounds__(256) score_kernel(const __bf16* __restrict__ dec_bf,
                                                    const __bf16* __restrict__ encp,
                                                    float* __restrict__ att) {
  __shared__ __bf16 ls[2][64 * 32];
  const int tid  = threadIdx.x;
  const int lane = tid & 31;
  const int wv   = tid >> 5;
  const int b    = blockIdx.x >> 5;              // 8 t-tiles * 4 bands
  const int rr   = blockIdx.x & 31;
  const int row0 = (rr >> 2) * 64;               // t
  const int col0 = (rr & 3) * 512 + wv * 64;     // e
  const __bf16* decB = dec_bf + (size_t)b * TD * NN;
  const __bf16* epB  = encp   + (size_t)b * TE * NN;

  v8f c[4][4] = {};
  fill_a_tile_bf16(ls[0], decB, NN, row0, 0, tid);
  __syncthreads();
  for (int k0 = 0; k0 < NN; k0 += 32) {
    const int cur = (k0 >> 5) & 1;
    if (k0 + 32 < NN) fill_a_tile_bf16(ls[cur ^ 1], decB, NN, row0, k0 + 32, tid);
    v16bf a0 = lds_a_frag(ls[cur], 0,  lane);
    v16bf a1 = lds_a_frag(ls[cur], 16, lane);
    v16bf a2 = lds_a_frag(ls[cur], 32, lane);
    v16bf a3 = lds_a_frag(ls[cur], 48, lane);
#pragma unroll
    for (int j = 0; j < 4; ++j) {
      v16bf bm = load_b_frag_bf16(epB, NN, col0 + 16 * j, k0, lane);
      c[0][j] = wmma_bf16(a0, bm, c[0][j]);
      c[1][j] = wmma_bf16(a1, bm, c[1][j]);
      c[2][j] = wmma_bf16(a2, bm, c[2][j]);
      c[3][j] = wmma_bf16(a3, bm, c[3][j]);
    }
    __syncthreads();
  }

  float* attB = att + (size_t)b * TD * TE;
#pragma unroll
  for (int mi = 0; mi < 4; ++mi)
#pragma unroll
    for (int j = 0; j < 4; ++j)
      store_f32_tile(c[mi][j], attB, TE, row0 + 16 * mi, col0 + 16 * j, lane);
}

// ---------------------------------------------------------------------------
// Kernel 3: softmax over e (2048) per (b,t) row; writes normalized f32 in
// place (output) and a bf16 copy for the context GEMM.
// ---------------------------------------------------------------------------
__global__ void __launch_bounds__(256) softmax_kernel(float* __restrict__ att,
                                                      __bf16* __restrict__ att_bf) {
  __shared__ float red[256];
  const int tid = threadIdx.x;
  float*  p  = att    + (size_t)blockIdx.x * TE;
  __bf16* pb = att_bf + (size_t)blockIdx.x * TE;

  float m = -3.402823466e+38f;
  for (int i = tid; i < TE; i += 256) m = fmaxf(m, p[i]);
  red[tid] = m;
  __syncthreads();
  for (int s = 128; s > 0; s >>= 1) {
    if (tid < s) red[tid] = fmaxf(red[tid], red[tid + s]);
    __syncthreads();
  }
  m = red[0];
  __syncthreads();

  float sum = 0.0f;
  for (int i = tid; i < TE; i += 256) {
    float e = __expf(p[i] - m);
    p[i] = e;
    sum += e;
  }
  red[tid] = sum;
  __syncthreads();
  for (int s = 128; s > 0; s >>= 1) {
    if (tid < s) red[tid] += red[tid + s];
    __syncthreads();
  }
  const float inv = 1.0f / red[0];
  for (int i = tid; i < TE; i += 256) {
    float v = p[i] * inv;
    p[i]  = v;
    pb[i] = (__bf16)v;
  }
}

// ---------------------------------------------------------------------------
// Kernel 4: context[b,t,n] = sum_e att[b,t,e] * enc_proj[b,e,n].
// Block = 64 t-rows x 512 n (full N). A = att_bf double-buffered in LDS,
// B = enc_proj^T (K=e contiguous).
// ---------------------------------------------------------------------------
__global__ void __launch_bounds__(256) context_kernel(const __bf16* __restrict__ att_bf,
                                                      const __bf16* __restrict__ encpT,
                                                      float* __restrict__ ctx) {
  __shared__ __bf16 ls[2][64 * 32];
  const int tid  = threadIdx.x;
  const int lane = tid & 31;
  const int wv   = tid >> 5;
  const int b    = blockIdx.x >> 3;          // 8 t-tiles per batch
  const int row0 = (blockIdx.x & 7) * 64;    // t
  const int col0 = wv * 64;                  // n
  const __bf16* attB = att_bf + (size_t)b * TD * TE;
  const __bf16* epT  = encpT  + (size_t)b * NN * TE;

  v8f c[4][4] = {};
  fill_a_tile_bf16(ls[0], attB, TE, row0, 0, tid);
  __syncthreads();
  for (int k0 = 0; k0 < TE; k0 += 32) {
    const int cur = (k0 >> 5) & 1;
    if (k0 + 32 < TE) {
      fill_a_tile_bf16(ls[cur ^ 1], attB, TE, row0, k0 + 32, tid);
      __builtin_prefetch(epT + (size_t)(col0 + (lane & 15)) * TE + k0 + 32, 0, 1);
    }
    v16bf a0 = lds_a_frag(ls[cur], 0,  lane);
    v16bf a1 = lds_a_frag(ls[cur], 16, lane);
    v16bf a2 = lds_a_frag(ls[cur], 32, lane);
    v16bf a3 = lds_a_frag(ls[cur], 48, lane);
#pragma unroll
    for (int j = 0; j < 4; ++j) {
      v16bf bm = load_b_frag_bf16(epT, TE, col0 + 16 * j, k0, lane);
      c[0][j] = wmma_bf16(a0, bm, c[0][j]);
      c[1][j] = wmma_bf16(a1, bm, c[1][j]);
      c[2][j] = wmma_bf16(a2, bm, c[2][j]);
      c[3][j] = wmma_bf16(a3, bm, c[3][j]);
    }
    __syncthreads();
  }

  float* ctxB = ctx + (size_t)b * TD * NN;
#pragma unroll
  for (int mi = 0; mi < 4; ++mi)
#pragma unroll
    for (int j = 0; j < 4; ++j)
      store_f32_tile(c[mi][j], ctxB, NN, row0 + 16 * mi, col0 + 16 * j, lane);
}

// ---------------------------------------------------------------------------
// Launch
// ---------------------------------------------------------------------------
extern "C" void kernel_launch(void* const* d_in, const int* in_sizes, int n_in,
                              void* d_out, int out_size, void* d_ws, size_t ws_size,
                              hipStream_t stream) {
  const float* enc  = (const float*)d_in[0];   // [32, 2048, 512] f32
  const float* dec  = (const float*)d_in[1];   // [32, 512, 512]  f32
  const float* W    = (const float*)d_in[2];   // [512, 512]      f32
  const float* bias = (const float*)d_in[3];   // [512]           f32

  float* out = (float*)d_out;
  float* ctx = out;                                  // [32, 512, 512]
  float* att = out + (size_t)NB * TD * NN;           // [32, 512, 2048]

  char* ws = (char*)d_ws;
  const size_t WT_BYTES    = (size_t)NN * DD * sizeof(__bf16);      // 0.5 MiB
  const size_t ENCP_BYTES  = (size_t)NB * TE * NN * sizeof(__bf16); // 64 MiB
  const size_t DECBF_BYTES = (size_t)NB * TD * NN * sizeof(__bf16); // 16 MiB
  __bf16* Wt     = (__bf16*)(ws);
  __bf16* encp   = (__bf16*)(ws + WT_BYTES);
  __bf16* encpT  = (__bf16*)(ws + WT_BYTES + ENCP_BYTES);
  __bf16* dec_bf = (__bf16*)(ws + WT_BYTES + 2 * ENCP_BYTES);
  __bf16* att_bf = (__bf16*)(ws + WT_BYTES + 2 * ENCP_BYTES + DECBF_BYTES);

  // 0a) W^T -> bf16 (K-contiguous B operand for the projection)
  wt_pack_kernel<<<(NN * DD) / 256, 256, 0, stream>>>(W, Wt);

  // 0b) dec -> bf16 (A operand of the score GEMM, converted once)
  dec_pack_kernel<<<(NB * TD * NN) / (256 * 4), 256, 0, stream>>>(dec, dec_bf);

  // 1) enc_proj = enc @ W + b : 32 batches * 32 e-tiles = 1024 blocks
  proj_kernel<<<NB * (TE / 64), 256, 0, stream>>>(enc, bias, Wt, encp, encpT);

  // 2) raw scores: 32 batches * 8 t-tiles * 4 e-bands = 1024 blocks
  score_kernel<<<NB * (TD / 64) * (TE / 512), 256, 0, stream>>>(dec_bf, encp, att);

  // 3) softmax (f32 in place + bf16 copy), one block per (b,t) row
  softmax_kernel<<<NB * TD, 256, 0, stream>>>(att, att_bf);

  // 4) context = attention @ enc_proj : 32 batches * 8 t-tiles = 256 blocks
  context_kernel<<<NB * (TD / 64), 256, 0, stream>>>(att_bf, encpT, ctx);
}
